// GCNEncoder_83863531421983
// MI455X (gfx1250) — matrix operand order repaired and checked by reading, not compiled, not run on hardware
//
#include <hip/hip_runtime.h>

#define N_NODES 50000
#define N_EDGES 800000
#define D_HID 128
#define D_LAT 64
#define BN_EPS 1e-5f

typedef __attribute__((ext_vector_type(8)))  __bf16 v8bf;
typedef __attribute__((ext_vector_type(16))) __bf16 v16bf;
typedef __attribute__((ext_vector_type(8)))  float  v8f;

__device__ __forceinline__ unsigned short f2bf(float f) {
    unsigned int u = __float_as_uint(f);
    u += 0x7FFFu + ((u >> 16) & 1u);   // round-to-nearest-even
    return (unsigned short)(u >> 16);
}

// ---------------- prep kernels ----------------
__global__ void prep_x_kernel(const float* __restrict__ x, unsigned short* __restrict__ xbf) {
    size_t idx = (size_t)blockIdx.x * blockDim.x + threadIdx.x;
    if (idx >= (size_t)N_NODES * D_HID) return;
    xbf[idx] = f2bf(x[idx]);
}

// Transpose + convert weights so B columns become contiguous rows (Wt[n][k]).
__global__ void prep_w_kernel(const float* __restrict__ W1, const float* __restrict__ Wmu,
                              const float* __restrict__ Wls,
                              unsigned short* __restrict__ w1t, unsigned short* __restrict__ wmut,
                              unsigned short* __restrict__ wlst) {
    int idx = blockIdx.x * blockDim.x + threadIdx.x;    // 0 .. 128*128-1
    int k = idx >> 7, n = idx & 127;
    w1t[n * 128 + k] = f2bf(W1[k * 128 + n]);
    if (n < 64) {
        wmut[n * 128 + k] = f2bf(Wmu[k * 64 + n]);
        wlst[n * 128 + k] = f2bf(Wls[k * 64 + n]);
    }
}

// ---------------- degree kernels ----------------
__global__ void deg_count_kernel(const long long* __restrict__ dst, float* __restrict__ deg) {
    int e = blockIdx.x * blockDim.x + threadIdx.x;
    if (e >= N_EDGES) return;
    atomicAdd(&deg[(int)dst[e]], 1.0f);
}

__global__ void deg_finalize_kernel(const float* __restrict__ deg, float* __restrict__ isd,
                                    float* __restrict__ invdeg) {
    int n = blockIdx.x * blockDim.x + threadIdx.x;
    if (n >= N_NODES) return;
    float d = deg[n] + 1.0f;          // +1 for self loop
    isd[n] = rsqrtf(d);
    invdeg[n] = 1.0f / d;
}

// ---------------- WMMA GEMM: C[M x N] = A[M x 128] * B[128 x N], Bt is B transposed ----------
// One wave computes one 16x16 tile via v_wmma_f32_16x16x32_bf16; block covers the N dimension.
__global__ void gemm_bf16_wmma(const unsigned short* __restrict__ A,
                               const unsigned short* __restrict__ Bt,
                               float* __restrict__ C, int N) {
    const int lane = threadIdx.x & 31;
    const int wave = threadIdx.x >> 5;
    const int m0 = blockIdx.x * 16;
    const int n0 = wave * 16;
    const int row = lane & 15;
    // A 16x32 bf16 fragment: lanes 0-15 hold K 0..15; lanes 16-31 hold K 16..31
    const int kA0 = (lane < 16) ? 0 : 16;
    // B 32x16 bf16 fragment: lanes 0-15 hold K 0..15 of col n; lanes 16-31 hold K 16..31
    const int kB0 = (lane < 16) ? 0 : 16;
    const unsigned short* arow = A + (size_t)(m0 + row) * D_HID;
    const unsigned short* brow = Bt + (size_t)(n0 + row) * D_HID;

    v8f acc = {};
#pragma unroll
    for (int k = 0; k < D_HID; k += 32) {
        v8bf alo = *reinterpret_cast<const v8bf*>(arow + k + kA0);
        v8bf ahi = *reinterpret_cast<const v8bf*>(arow + k + kA0 + 8);
        v16bf a;
#pragma unroll
        for (int i = 0; i < 8; ++i) { a[i] = alo[i]; a[i + 8] = ahi[i]; }
        v16bf b = *reinterpret_cast<const v16bf*>(brow + k + kB0);
        acc = __builtin_amdgcn_wmma_f32_16x16x32_bf16(false, a, false, b, (short)0, acc,
                                                      false, false);
    }
    // C/D layout: VGPR r = row (m0 + r) for lanes 0-15, (m0 + 8 + r) for lanes 16-31; col = lane&15
    const int mbase = m0 + ((lane < 16) ? 0 : 8);
    float* crow = C + (size_t)mbase * N + n0 + (lane & 15);
#pragma unroll
    for (int r = 0; r < 8; ++r) crow[(size_t)r * N] = acc[r];
}

// ---------------- edge scatter, 128 features (one edge per wave, float4 per lane) -----------
__global__ void scatter128_kernel(const float* __restrict__ h0, const long long* __restrict__ src,
                                  const long long* __restrict__ dst, const float* __restrict__ isd,
                                  float* __restrict__ agg) {
    int e = blockIdx.x * 8 + (threadIdx.x >> 5);
    if (e >= N_EDGES) return;
    int lane = threadIdx.x & 31;
    int s = (int)src[e], d = (int)dst[e];
    float ne = isd[s] * isd[d];
    float4 v = reinterpret_cast<const float4*>(h0 + (size_t)s * 128)[lane];
    float* base = agg + (size_t)d * 128 + lane * 4;
    atomicAdd(base + 0, v.x * ne);
    atomicAdd(base + 1, v.y * ne);
    atomicAdd(base + 2, v.z * ne);
    atomicAdd(base + 3, v.w * ne);
}

// ---------------- combine self-loop + bias, accumulate BN statistics ----------------
__global__ void combine_bn_stats_kernel(float* __restrict__ hpre /* in: agg, out: pre-BN */,
                                        const float* __restrict__ h0,
                                        const float* __restrict__ invdeg,
                                        const float* __restrict__ b1,
                                        float* __restrict__ gsum, float* __restrict__ gsq) {
    int f = threadIdx.x;               // 128 threads: one feature each
    int n0 = blockIdx.x * 50;          // 50 nodes per block, 1000 blocks
    float s = 0.f, q = 0.f;
    float bf = b1[f];
    for (int i = 0; i < 50; ++i) {
        int n = n0 + i;
        size_t idx = (size_t)n * 128 + f;
        float v = hpre[idx] + h0[idx] * invdeg[n] + bf;
        hpre[idx] = v;
        s += v;
        q += v * v;
    }
    atomicAdd(&gsum[f], s);
    atomicAdd(&gsq[f], q);
}

// ---------------- BatchNorm + ReLU, emit bf16 activations ----------------
__global__ void bn_relu_kernel(const float* __restrict__ hpre, const float* __restrict__ gsum,
                               const float* __restrict__ gsq, const float* __restrict__ gamma,
                               const float* __restrict__ beta, unsigned short* __restrict__ hbf) {
    size_t idx = (size_t)blockIdx.x * blockDim.x + threadIdx.x;
    if (idx >= (size_t)N_NODES * 128) return;
    int f = idx & 127;
    const float invN = 1.0f / (float)N_NODES;
    float mean = gsum[f] * invN;
    float var = gsq[f] * invN - mean * mean;
    float v = (hpre[idx] - mean) * rsqrtf(var + BN_EPS) * gamma[f] + beta[f];
    v = fmaxf(v, 0.0f);
    hbf[idx] = f2bf(v);
}

// ---------------- edge scatter, 64 features, mu + log_std in one pass ----------------
__global__ void scatter64x2_kernel(const float* __restrict__ hm0, const float* __restrict__ hl0,
                                   const long long* __restrict__ src,
                                   const long long* __restrict__ dst,
                                   const float* __restrict__ isd, float* __restrict__ aggm,
                                   float* __restrict__ aggl) {
    int e = blockIdx.x * 8 + (threadIdx.x >> 5);
    if (e >= N_EDGES) return;
    int lane = threadIdx.x & 31;
    int s = (int)src[e], d = (int)dst[e];
    float ne = isd[s] * isd[d];
    float2 vm = reinterpret_cast<const float2*>(hm0 + (size_t)s * 64)[lane];
    float2 vl = reinterpret_cast<const float2*>(hl0 + (size_t)s * 64)[lane];
    size_t o = (size_t)d * 64 + lane * 2;
    atomicAdd(&aggm[o + 0], vm.x * ne);
    atomicAdd(&aggm[o + 1], vm.y * ne);
    atomicAdd(&aggl[o + 0], vl.x * ne);
    atomicAdd(&aggl[o + 1], vl.y * ne);
}

// ---------------- final combine into d_out (mu then log_std) ----------------
__global__ void finalize_out_kernel(const float* __restrict__ aggm, const float* __restrict__ aggl,
                                    const float* __restrict__ hm0, const float* __restrict__ hl0,
                                    const float* __restrict__ invdeg, const float* __restrict__ bmu,
                                    const float* __restrict__ bls, float* __restrict__ out) {
    size_t idx = (size_t)blockIdx.x * blockDim.x + threadIdx.x;
    const size_t TOT = (size_t)N_NODES * 64;
    if (idx >= TOT) return;
    int n = (int)(idx >> 6), f = (int)(idx & 63);
    float id = invdeg[n];
    out[idx] = aggm[idx] + hm0[idx] * id + bmu[f];
    out[TOT + idx] = aggl[idx] + hl0[idx] * id + bls[f];
}

extern "C" void kernel_launch(void* const* d_in, const int* in_sizes, int n_in, void* d_out,
                              int out_size, void* d_ws, size_t ws_size, hipStream_t stream) {
    (void)in_sizes; (void)n_in; (void)out_size; (void)ws_size;
    const float* x = (const float*)d_in[0];
    const long long* ei = (const long long*)d_in[1];   // int64 [2, E]
    const float* W1 = (const float*)d_in[2];
    const float* b1 = (const float*)d_in[3];
    const float* gamma = (const float*)d_in[4];
    const float* beta = (const float*)d_in[5];
    const float* Wmu = (const float*)d_in[6];
    const float* bmu = (const float*)d_in[7];
    const float* Wls = (const float*)d_in[8];
    const float* bls = (const float*)d_in[9];
    const long long* src = ei;
    const long long* dst = ei + N_EDGES;

    char* p = (char*)d_ws;
    auto alloc = [&](size_t bytes) -> char* {
        char* r = p;
        p += (bytes + 255) & ~(size_t)255;
        return r;
    };
    float* deg = (float*)alloc((size_t)N_NODES * 4);
    float* isd = (float*)alloc((size_t)N_NODES * 4);
    float* invdeg = (float*)alloc((size_t)N_NODES * 4);
    unsigned short* xbf = (unsigned short*)alloc((size_t)N_NODES * 128 * 2); // reused as h-bf16
    unsigned short* w1t = (unsigned short*)alloc(128 * 128 * 2);
    unsigned short* wmut = (unsigned short*)alloc(64 * 128 * 2);
    unsigned short* wlst = (unsigned short*)alloc(64 * 128 * 2);
    float* h0 = (float*)alloc((size_t)N_NODES * 128 * 4);
    float* agg1 = (float*)alloc((size_t)N_NODES * 128 * 4);  // becomes pre-BN h in-place
    float* hm0 = (float*)alloc((size_t)N_NODES * 64 * 4);
    float* hl0 = (float*)alloc((size_t)N_NODES * 64 * 4);
    float* aggm = (float*)alloc((size_t)N_NODES * 64 * 4);
    float* aggl = (float*)alloc((size_t)N_NODES * 64 * 4);
    float* gsum = (float*)alloc(128 * 4);
    float* gsq = (float*)alloc(128 * 4);

    hipMemsetAsync(deg, 0, (size_t)N_NODES * 4, stream);
    hipMemsetAsync(agg1, 0, (size_t)N_NODES * 128 * 4, stream);
    hipMemsetAsync(aggm, 0, (size_t)N_NODES * 64 * 4, stream);
    hipMemsetAsync(aggl, 0, (size_t)N_NODES * 64 * 4, stream);
    hipMemsetAsync(gsum, 0, 128 * 4, stream);
    hipMemsetAsync(gsq, 0, 128 * 4, stream);

    prep_x_kernel<<<(N_NODES * 128) / 256, 256, 0, stream>>>(x, xbf);
    prep_w_kernel<<<64, 256, 0, stream>>>(W1, Wmu, Wls, w1t, wmut, wlst);
    deg_count_kernel<<<(N_EDGES + 255) / 256, 256, 0, stream>>>(dst, deg);
    deg_finalize_kernel<<<(N_NODES + 255) / 256, 256, 0, stream>>>(deg, isd, invdeg);

    // Layer 1: h0 = x @ W1 (bf16 WMMA, f32 accumulate), N = 128 -> 8 waves/block
    gemm_bf16_wmma<<<N_NODES / 16, 256, 0, stream>>>(xbf, w1t, h0, 128);
    scatter128_kernel<<<N_EDGES / 8, 256, 0, stream>>>(h0, src, dst, isd, agg1);
    combine_bn_stats_kernel<<<N_NODES / 50, 128, 0, stream>>>(agg1, h0, invdeg, b1, gsum, gsq);
    bn_relu_kernel<<<(N_NODES * 128) / 256, 256, 0, stream>>>(agg1, gsum, gsq, gamma, beta, xbf);

    // Layer 2: two N=64 GEMMs -> 4 waves/block each
    gemm_bf16_wmma<<<N_NODES / 16, 128, 0, stream>>>(xbf, wmut, hm0, 64);
    gemm_bf16_wmma<<<N_NODES / 16, 128, 0, stream>>>(xbf, wlst, hl0, 64);
    scatter64x2_kernel<<<N_EDGES / 8, 256, 0, stream>>>(hm0, hl0, src, dst, isd, aggm, aggl);
    finalize_out_kernel<<<((size_t)N_NODES * 64 + 255) / 256, 256, 0, stream>>>(
        aggm, aggl, hm0, hl0, invdeg, bmu, bls, (float*)d_out);
}
